// GATRegressor_47193100649160
// MI455X (gfx1250) — matrix operand order repaired and checked by reading, not compiled, not run on hardware
//
#include <hip/hip_runtime.h>
#include <hip/hip_bf16.h>

#define NN 50000       // nodes
#define NE 800000      // real edges
#define EALL 850000    // edges + self loops
#define EDIM 32        // edge feature dim
#define INDIM 128
#define HIDC 64        // per-head channels (both layers)
#define H1 4           // heads layer 1
#define HC1 256        // H1*HIDC
#define NG 64          // graphs
#define NEG_SLOPE_C 0.2f

typedef __attribute__((ext_vector_type(16))) _Float16 v16h;
typedef __attribute__((ext_vector_type(8)))  float    v8f;

// ---------------- utility ----------------

__global__ void fill_f32(float* __restrict__ p, float v, long n) {
    long t = (long)blockIdx.x * blockDim.x + threadIdx.x;
    if (t < n) p[t] = v;
}

__device__ __forceinline__ void atomicMaxFloat(float* addr, float val) {
    // valid with -inf init: positives via signed-int max, negatives via unsigned min
    if (val >= 0.0f) atomicMax((int*)addr, __float_as_int(val));
    else             atomicMin((unsigned int*)addr, __float_as_uint(val));
}

// ---------------- self-loop attr = mean of incoming edge attrs ----------------

__global__ void loop_sum_kernel(const int* __restrict__ ei, const float* __restrict__ eattr,
                                float* __restrict__ lsum, float* __restrict__ lcnt) {
    long t = (long)blockIdx.x * blockDim.x + threadIdx.x;
    if (t >= (long)NE * EDIM) return;
    int e = (int)(t >> 5), k = (int)(t & 31);
    int dst = ei[NE + e];
    atomicAdd(&lsum[(long)dst * EDIM + k], eattr[t]);
    if (k == 0) atomicAdd(&lcnt[dst], 1.0f);
}

__global__ void loop_div_kernel(float* __restrict__ lsum, const float* __restrict__ lcnt) {
    long t = (long)blockIdx.x * blockDim.x + threadIdx.x;
    if (t >= (long)NN * EDIM) return;
    lsum[t] /= fmaxf(lcnt[t >> 5], 1.0f);
}

// ---------------- WMMA GEMM: C[M,N] = A[M,K] @ B[K,N], f32 in/out, f16 MACs ----------------
// One wave computes a 16 x (16*NT) strip: A fragment loaded once per k-step,
// NT back-to-back WMMAs against NT B fragments. Requires M%16==0, N%(16*NT)==0, K%32==0.

template <int NT>
__global__ __launch_bounds__(256) void gemm_wmma_strip(const float* __restrict__ A,
                                                       const float* __restrict__ B,
                                                       float* __restrict__ C,
                                                       int M, int N, int K) {
    int wave = blockIdx.x * (blockDim.x >> 5) + (threadIdx.x >> 5);
    int lane = threadIdx.x & 31;
    int strips = N / (16 * NT);
    int total  = (M >> 4) * strips;
    if (wave >= total) return;               // wave-uniform: EXEC stays all-ones
    int mt = wave / strips, st = wave % strips;
    int g = lane >> 4, r = lane & 15;
    long arow  = (long)(mt * 16 + r) * K;    // A row owned by this lane
    int  bcol0 = st * (16 * NT) + r;         // first B column owned by this lane

    v8f zero = {};
    v8f acc[NT];
#pragma unroll
    for (int t = 0; t < NT; ++t) acc[t] = zero;

    for (int kt = 0; kt < K; kt += 32) {
        if (kt + 32 < K) __builtin_prefetch(A + arow + kt + 32, 0, 1);
        v16h a;
        // A 16x32 f16 fragment: VGPR v halves -> k = (v<4 ? 2v : 16+2(v-4)) + 8*grp
#pragma unroll
        for (int v = 0; v < 8; ++v) {
            int kb = ((v < 4) ? (2 * v) : (16 + 2 * (v - 4))) + g * 8 + kt;
            const float2 f = *(const float2*)(A + arow + kb);
            a[2 * v]     = (_Float16)f.x;
            a[2 * v + 1] = (_Float16)f.y;
        }
#pragma unroll
        for (int t = 0; t < NT; ++t) {
            // B 32x16 f16 fragment: lane holds column; element e -> k = (e<8 ? e : 8+e) + 8*grp
            v16h b;
#pragma unroll
            for (int e = 0; e < 16; ++e) {
                int kb = ((e < 8) ? e : (e + 8)) + g * 8 + kt;
                b[e] = (_Float16)B[(long)kb * N + bcol0 + t * 16];
            }
            acc[t] = __builtin_amdgcn_wmma_f32_16x16x32_f16(false, a, false, b,
                                                            (short)0, acc[t], false, false);
        }
    }
    // D: vgpr i -> m = 8*grp + i, n = lane&15 (+ 16*t)
#pragma unroll
    for (int t = 0; t < NT; ++t)
#pragma unroll
        for (int i = 0; i < 8; ++i)
            C[(long)(mt * 16 + g * 8 + i) * N + bcol0 + t * 16] = acc[t][i];
}

// ---------------- attention pre-reductions ----------------

// Be[k,h] = sum_c We[k, h*C + c] * ae[h, c]     (folds he=e@We out of existence)
__global__ void precompute_be(const float* __restrict__ We, const float* __restrict__ ae,
                              float* __restrict__ Be, int H, int HC) {
    int t = blockIdx.x * blockDim.x + threadIdx.x;
    if (t >= EDIM * H) return;
    int k = t / H, h = t % H;
    float s = 0.0f;
    for (int c = 0; c < HIDC; ++c) s += We[(long)k * HC + h * HIDC + c] * ae[h * HIDC + c];
    Be[k * H + h] = s;
}

// al_src[n,h], al_dst[n,h] from h-features
__global__ void node_attn(const float* __restrict__ hfeat, const float* __restrict__ a_src,
                          const float* __restrict__ a_dst, float* __restrict__ als,
                          float* __restrict__ ald, int H) {
    int t = blockIdx.x * blockDim.x + threadIdx.x;
    if (t >= NN * H) return;
    int n = t / H, h = t % H;
    const float* hp = hfeat + (long)n * H * HIDC + h * HIDC;
    float s = 0.0f, d = 0.0f;
    for (int c = 0; c < HIDC; ++c) {
        float v = hp[c];
        s += v * a_src[h * HIDC + c];
        d += v * a_dst[h * HIDC + c];
    }
    als[t] = s;
    ald[t] = d;
}

// ---------------- edge softmax ----------------

__global__ void edge_logits_max(const int* __restrict__ ei, const float* __restrict__ eattr,
                                const float* __restrict__ lattr, const float* __restrict__ als,
                                const float* __restrict__ ald, const float* __restrict__ Be,
                                float* __restrict__ logits, float* __restrict__ mbuf, int H) {
    int e = blockIdx.x * blockDim.x + threadIdx.x;
    if (e >= EALL) return;
    int src, dst; const float* ap;
    if (e < NE) { src = ei[e]; dst = ei[NE + e]; ap = eattr + (long)e * EDIM; }
    else        { src = dst = e - NE;            ap = lattr + (long)(e - NE) * EDIM; }
    for (int h = 0; h < H; ++h) {
        float ale = 0.0f;
#pragma unroll
        for (int k = 0; k < EDIM; ++k) ale += ap[k] * Be[k * H + h];
        float l = als[src * H + h] + ald[dst * H + h] + ale;
        l = (l > 0.0f) ? l : NEG_SLOPE_C * l;
        logits[(long)e * H + h] = l;
        atomicMaxFloat(&mbuf[dst * H + h], l);
    }
}

__global__ void edge_exp_sum(const int* __restrict__ ei, const float* __restrict__ logits,
                             const float* __restrict__ mbuf, float* __restrict__ dbuf, int H) {
    long t = (long)blockIdx.x * blockDim.x + threadIdx.x;
    if (t >= (long)EALL * H) return;
    int e = (int)(t / H), h = (int)(t % H);
    int dst = (e < NE) ? ei[NE + e] : (e - NE);
    float m = mbuf[dst * H + h];
    if (m < -3.0e38f) m = 0.0f;  // isfinite() clamp (m init = -inf)
    atomicAdd(&dbuf[dst * H + h], __expf(logits[t] - m));
}

// wave-per-edge weighted scatter: out[dst,h,c] += h[src,h,c] * alpha[e,h]
__global__ void aggregate(const int* __restrict__ ei, const float* __restrict__ hfeat,
                          const float* __restrict__ logits, const float* __restrict__ mbuf,
                          const float* __restrict__ dbuf, float* __restrict__ out,
                          int H, int HC) {
    int wave = (int)(((long)blockIdx.x * blockDim.x + threadIdx.x) >> 5);
    int lane = threadIdx.x & 31;
    if (wave >= EALL) return;
    int e = wave, src, dst;
    if (e < NE) { src = ei[e]; dst = ei[NE + e]; } else { src = dst = e - NE; }
    for (int j = lane; j < HC; j += 32) {
        int h = j >> 6;  // HIDC == 64
        float m = mbuf[dst * H + h];
        if (m < -3.0e38f) m = 0.0f;
        float alpha = __expf(logits[(long)e * H + h] - m) / (dbuf[dst * H + h] + 1e-16f);
        atomicAdd(&out[(long)dst * HC + j], hfeat[(long)src * HC + j] * alpha);
    }
}

__global__ void bias_act(float* __restrict__ x, const float* __restrict__ b, int F, int relu) {
    long t = (long)blockIdx.x * blockDim.x + threadIdx.x;
    if (t >= (long)NN * F) return;
    float v = x[t] + b[t % F];
    x[t] = relu ? fmaxf(v, 0.0f) : v;
}

// ---------------- pooling + FC ----------------

__global__ void pool_kernel(const float* __restrict__ h2, const int* __restrict__ batch,
                            float* __restrict__ pooled, float* __restrict__ cntg) {
    long t = (long)blockIdx.x * blockDim.x + threadIdx.x;
    if (t >= (long)NN * HIDC) return;
    int n = (int)(t >> 6), c = (int)(t & 63);
    int g = batch[n];
    atomicAdd(&pooled[(long)g * HIDC + c], h2[t]);
    if (c == 0) atomicAdd(&cntg[g], 1.0f);
}

__global__ void final_fc(const float* __restrict__ pooled, const float* __restrict__ cntg,
                         const float* __restrict__ Wfc, const float* __restrict__ bfc,
                         float* __restrict__ out) {
    int g = blockIdx.x * blockDim.x + threadIdx.x;
    if (g >= NG) return;
    float c = fmaxf(cntg[g], 1.0f);
    float s = 0.0f;
    for (int j = 0; j < HIDC; ++j) s += (pooled[(long)g * HIDC + j] / c) * Wfc[j];
    out[g] = s + bfc[0];
}

// ---------------- launch ----------------

static inline int blk(long n) { return (int)((n + 255) / 256); }

extern "C" void kernel_launch(void* const* d_in, const int* in_sizes, int n_in,
                              void* d_out, int out_size, void* d_ws, size_t ws_size,
                              hipStream_t stream) {
    const float* x     = (const float*)d_in[0];
    const int*   ei    = (const int*)  d_in[1];   // [2, NE]
    const float* eattr = (const float*)d_in[2];   // [NE, 32]
    const int*   batch = (const int*)  d_in[3];   // [NN]
    const float* W1  = (const float*)d_in[4];
    const float* We1 = (const float*)d_in[5];
    const float* as1 = (const float*)d_in[6];
    const float* ad1 = (const float*)d_in[7];
    const float* ae1 = (const float*)d_in[8];
    const float* b1  = (const float*)d_in[9];
    const float* W2  = (const float*)d_in[10];
    const float* We2 = (const float*)d_in[11];
    const float* as2 = (const float*)d_in[12];
    const float* ad2 = (const float*)d_in[13];
    const float* ae2 = (const float*)d_in[14];
    const float* b2  = (const float*)d_in[15];
    const float* Wfc = (const float*)d_in[16];
    const float* bfc = (const float*)d_in[17];
    float* out = (float*)d_out;

    // workspace carve-up (floats)
    float* ws = (float*)d_ws;
    long off = 0;
    float* lsum   = ws + off; off += (long)NN * EDIM;   // loop-attr sums -> means (in place)
    float* lcnt   = ws + off; off += 50048;
    float* h1     = ws + off; off += (long)NN * HC1;    // x@W1
    float* out1   = ws + off; off += (long)NN * HC1;    // layer1 aggregate -> relu'd act
    float* als1   = ws + off; off += (long)NN * H1;
    float* ald1   = ws + off; off += (long)NN * H1;
    float* m1     = ws + off; off += (long)NN * H1;
    float* dd1    = ws + off; off += (long)NN * H1;
    float* logit1 = ws + off; off += (long)EALL * H1;
    float* Be1    = ws + off; off += 128;
    float* h2     = ws + off; off += (long)NN * HIDC;   // out1@W2
    float* out2   = ws + off; off += (long)NN * HIDC;
    float* als2   = ws + off; off += (long)NN;
    float* ald2   = ws + off; off += (long)NN;
    float* m2     = ws + off; off += (long)NN;
    float* dd2    = ws + off; off += (long)NN;
    float* logit2 = ws + off; off += (long)EALL;
    float* Be2    = ws + off; off += 32;
    float* pooled = ws + off; off += (long)NG * HIDC;
    float* cntg   = ws + off; off += NG;
    (void)ws_size; (void)n_in; (void)in_sizes; (void)out_size;

    // re-initialize accumulators every call (graph replays do not re-poison)
    fill_f32<<<blk((long)NN * EDIM), 256, 0, stream>>>(lsum, 0.0f, (long)NN * EDIM);
    fill_f32<<<blk(NN), 256, 0, stream>>>(lcnt, 0.0f, NN);
    fill_f32<<<blk((long)NN * HC1), 256, 0, stream>>>(out1, 0.0f, (long)NN * HC1);
    fill_f32<<<blk((long)NN * H1), 256, 0, stream>>>(dd1, 0.0f, (long)NN * H1);
    fill_f32<<<blk((long)NN * H1), 256, 0, stream>>>(m1, -__builtin_inff(), (long)NN * H1);
    fill_f32<<<blk((long)NN * HIDC), 256, 0, stream>>>(out2, 0.0f, (long)NN * HIDC);
    fill_f32<<<blk(NN), 256, 0, stream>>>(dd2, 0.0f, NN);
    fill_f32<<<blk(NN), 256, 0, stream>>>(m2, -__builtin_inff(), NN);
    fill_f32<<<blk((long)NG * HIDC), 256, 0, stream>>>(pooled, 0.0f, (long)NG * HIDC);
    fill_f32<<<blk(NG), 256, 0, stream>>>(cntg, 0.0f, NG);

    // self-loop attrs (mean of incoming)
    loop_sum_kernel<<<blk((long)NE * EDIM), 256, 0, stream>>>(ei, eattr, lsum, lcnt);
    loop_div_kernel<<<blk((long)NN * EDIM), 256, 0, stream>>>(lsum, lcnt);

    // ---- layer 1 ----
    gemm_wmma_strip<4><<<((NN / 16) * (HC1 / 64) + 7) / 8, 256, 0, stream>>>(x, W1, h1, NN, HC1, INDIM);
    precompute_be<<<1, EDIM * H1, 0, stream>>>(We1, ae1, Be1, H1, HC1);
    node_attn<<<blk((long)NN * H1), 256, 0, stream>>>(h1, as1, ad1, als1, ald1, H1);
    edge_logits_max<<<blk(EALL), 256, 0, stream>>>(ei, eattr, lsum, als1, ald1, Be1, logit1, m1, H1);
    edge_exp_sum<<<blk((long)EALL * H1), 256, 0, stream>>>(ei, logit1, m1, dd1, H1);
    aggregate<<<blk((long)EALL * 32), 256, 0, stream>>>(ei, h1, logit1, m1, dd1, out1, H1, HC1);
    bias_act<<<blk((long)NN * HC1), 256, 0, stream>>>(out1, b1, HC1, 1);

    // ---- layer 2 (heads=1, mean == identity) ----
    gemm_wmma_strip<4><<<((NN / 16) * (HIDC / 64) + 7) / 8, 256, 0, stream>>>(out1, W2, h2, NN, HIDC, HC1);
    precompute_be<<<1, EDIM, 0, stream>>>(We2, ae2, Be2, 1, HIDC);
    node_attn<<<blk(NN), 256, 0, stream>>>(h2, as2, ad2, als2, ald2, 1);
    edge_logits_max<<<blk(EALL), 256, 0, stream>>>(ei, eattr, lsum, als2, ald2, Be2, logit2, m2, 1);
    edge_exp_sum<<<blk(EALL), 256, 0, stream>>>(ei, logit2, m2, dd2, 1);
    aggregate<<<blk((long)EALL * 32), 256, 0, stream>>>(ei, h2, logit2, m2, dd2, out2, 1, HIDC);
    bias_act<<<blk((long)NN * HIDC), 256, 0, stream>>>(out2, b2, HIDC, 0);

    // ---- pool + FC ----
    pool_kernel<<<blk((long)NN * HIDC), 256, 0, stream>>>(out2, batch, pooled, cntg);
    final_fc<<<1, NG, 0, stream>>>(pooled, cntg, Wfc, bfc, out);
}
